// Model_29171417874946
// MI455X (gfx1250) — compile-verified, matrix operand order checked by reading, not run on hardware
//
#include <hip/hip_runtime.h>

// ---------------------------------------------------------------------------
// Problem constants (from the reference)
// ---------------------------------------------------------------------------
#define N_SRC0      400000
#define N_DST0      200000
#define N_DST1      100000
#define E0          3200000
#define E1          1600000
#define IN_FEATS    128
#define H_FEATS     256
#define NUM_CLASSES 8
#define XROW        (IN_FEATS + 1)   // 129: col0 = cell id
#define K1          256              // concat K for layer-1 GEMM
#define N2          16               // concat N for layer-2 GEMM (8 self | 8 neigh)

// LDS staging for Wt1 in gemm1: 256 rows, padded to 264 bf16 (528 B) per row
// so lane->bank stride is 132 words == 4 (mod 64): halves are conflict-light.
#define BROW_PAD    264
#define LDS1_BYTES  (K1 * BROW_PAD * 2)   // 135168 B (<320KB/WGP; 2 blocks/WGP)

#define AS3 __attribute__((address_space(3)))
#define AS1 __attribute__((address_space(1)))

#if defined(__has_builtin)
#  if __has_builtin(__builtin_amdgcn_global_load_async_to_lds_b128)
#    define HAVE_ASYNC_TO_LDS 1
#  endif
#endif

// ---------------------------------------------------------------------------
// WMMA vector types (CDNA5 wave32)
// ---------------------------------------------------------------------------
typedef __attribute__((ext_vector_type(16))) __bf16 v16bf;
typedef __attribute__((ext_vector_type(8)))  __bf16 v8bf;
typedef __attribute__((ext_vector_type(8)))  float  v8f;
typedef __attribute__((ext_vector_type(4)))  int    v4i;
typedef AS3 __bf16 lds_bf16;

__device__ __forceinline__ __bf16 f32_to_bf16(float f) {
    unsigned int u = __builtin_bit_cast(unsigned int, f);
    unsigned int r = u + 0x7FFFu + ((u >> 16) & 1u);   // round-to-nearest-even
    unsigned short h = (unsigned short)(r >> 16);
    return __builtin_bit_cast(__bf16, h);
}

// Fragment half-pair loader for v_wmma_f32_16x16x32_bf16. Per the documented
// wave32 layout a lane holds K = [k0..k0+7] and [k0+16..k0+23], k0=8*(lane>=16);
// with K-contiguous storage that is two aligned 16B loads.
__device__ __forceinline__ v16bf load_frag16(const __bf16* __restrict__ p) {
    v8bf lo = *(const v8bf*)(p);
    v8bf hi = *(const v8bf*)(p + 16);
    v16bf f;
#pragma unroll
    for (int i = 0; i < 8; ++i) { f[i] = lo[i]; f[i + 8] = hi[i]; }
    return f;
}

__device__ __forceinline__ v16bf load_frag16_lds(const lds_bf16* p) {
    v8bf lo = *(const AS3 v8bf*)(p);
    v8bf hi = *(const AS3 v8bf*)(p + 16);
    v16bf f;
#pragma unroll
    for (int i = 0; i < 8; ++i) { f[i] = lo[i]; f[i + 8] = hi[i]; }
    return f;
}

// ---------------------------------------------------------------------------
// 1) Gate: h[i] = feat[i] * sigmoid(emb[cell_id[i]])   (32 lanes per node)
// ---------------------------------------------------------------------------
__global__ void gate_kernel(const float* __restrict__ x,
                            const float* __restrict__ emb,
                            float* __restrict__ h) {
    unsigned t = blockIdx.x * blockDim.x + threadIdx.x;
    int node = t >> 5, lane = t & 31;
    if (node >= N_SRC0) return;
    int cid = (int)x[(size_t)node * XROW];
    const float* frow = x + (size_t)node * XROW + 1;
    const float* erow = emb + (size_t)cid * IN_FEATS;
    float* hrow = h + (size_t)node * IN_FEATS;
#pragma unroll
    for (int j = 0; j < 4; ++j) {
        int c = lane + j * 32;
        float s = 1.0f / (1.0f + __expf(-erow[c]));
        hrow[c] = frow[c] * s;
    }
}

// ---------------------------------------------------------------------------
// 2) Layer-1 edge scatter: agg0[dst] += h[src] (128 f32 atomics/edge), deg0++
// ---------------------------------------------------------------------------
__global__ void edge1_kernel(const int* __restrict__ src,
                             const int* __restrict__ dst,
                             const float* __restrict__ h,
                             float* __restrict__ agg,
                             float* __restrict__ deg) {
    unsigned t = blockIdx.x * blockDim.x + threadIdx.x;
    int e = t >> 5, lane = t & 31;
    if (e >= E0) return;
    int s = src[e], d = dst[e];
    const float* hs = h + (size_t)s * IN_FEATS;
    float* ad = agg + (size_t)d * IN_FEATS;
    if (lane == 0) atomicAdd(&deg[d], 1.0f);
#pragma unroll
    for (int j = 0; j < 4; ++j) {
        int c = lane + j * 32;
        atomicAdd(&ad[c], hs[c]);
    }
}

// ---------------------------------------------------------------------------
// 3) Build layer-1 GEMM operand A1 (bf16): row n = [ h[n] | agg0[n]/deg ]
// ---------------------------------------------------------------------------
__global__ void build_a1_kernel(const float* __restrict__ h,
                                const float* __restrict__ agg,
                                const float* __restrict__ deg,
                                __bf16* __restrict__ A1) {
    unsigned t = blockIdx.x * blockDim.x + threadIdx.x;
    int node = t >> 5, lane = t & 31;
    if (node >= N_DST0) return;
    float inv = 1.0f / fmaxf(deg[node], 1.0f);
    __bf16* arow = A1 + (size_t)node * K1;
    const float* hrow = h + (size_t)node * IN_FEATS;
    const float* grow = agg + (size_t)node * IN_FEATS;
#pragma unroll
    for (int j = 0; j < 4; ++j) {
        int c = lane + j * 32;
        arow[c]            = f32_to_bf16(hrow[c]);
        arow[IN_FEATS + c] = f32_to_bf16(grow[c] * inv);
    }
}

// ---------------------------------------------------------------------------
// Weight prep: transpose+concat to N-major bf16 so B fragments are b128 loads.
// Wt1[n][k] : k<128 -> W_self1[k][n], else W_neigh1[k-128][n]   (256 x 256)
// Wt2[c][k] : c<8  -> W_self2[k][c],  else W_neigh2[k][c-8]     (16  x 256)
// ---------------------------------------------------------------------------
__global__ void prep_w1_kernel(const float* __restrict__ Ws,
                               const float* __restrict__ Wn,
                               __bf16* __restrict__ Wt1) {
    unsigned t = blockIdx.x * blockDim.x + threadIdx.x;  // 256*256
    int n = t >> 8, k = t & 255;
    float v = (k < IN_FEATS) ? Ws[(size_t)k * H_FEATS + n]
                             : Wn[(size_t)(k - IN_FEATS) * H_FEATS + n];
    Wt1[(size_t)n * K1 + k] = f32_to_bf16(v);
}

__global__ void prep_w2_kernel(const float* __restrict__ Ws,
                               const float* __restrict__ Wn,
                               __bf16* __restrict__ Wt2) {
    unsigned t = blockIdx.x * blockDim.x + threadIdx.x;  // 16*256
    int c = t >> 8, k = t & 255;
    float v = (c < NUM_CLASSES) ? Ws[(size_t)k * NUM_CLASSES + c]
                                : Wn[(size_t)k * NUM_CLASSES + (c - NUM_CLASSES)];
    Wt2[(size_t)c * H_FEATS + k] = f32_to_bf16(v);
}

// ---------------------------------------------------------------------------
// 4) Layer-1 GEMM: h1 = relu(A1[200000,256] @ Wt1^T + b1) -> bf16 [200000,256]
//    8 waves/block: 2(M) x 4(N); wave tile 16(M) x 64(N) = 4 wmma accumulators.
//    Wt1 (128KB) is staged once per block into LDS (async-to-LDS when
//    available), so the K loop feeds WMMA from ds_load_b128 while the VMEM
//    pipe streams only the A matrix.
// ---------------------------------------------------------------------------
__global__ void gemm1_kernel(const __bf16* __restrict__ A,
                             const __bf16* __restrict__ Bt,
                             const float* __restrict__ b1,
                             __bf16* __restrict__ h1) {
    extern __shared__ char smem[];                 // LDS1_BYTES
    lds_bf16* Bs = (lds_bf16*)smem;

    // ---- cooperative stage: Wt1 [256 x 512B] -> LDS [256 x 528B padded] ----
    {
        const char* gsrc = (const char*)Bt;
#pragma unroll 4
        for (int c = threadIdx.x; c < (K1 * H_FEATS * 2) / 16; c += 256) {
            int n    = c >> 5;            // row (output col), 32 x16B per row
            int k16  = c & 31;
            int goff = n * (K1 * 2) + k16 * 16;
            int loff = n * (BROW_PAD * 2) + k16 * 16;
#if defined(HAVE_ASYNC_TO_LDS)
            __builtin_amdgcn_global_load_async_to_lds_b128(
                (AS1 v4i*)(gsrc + goff),
                (AS3 v4i*)((AS3 char*)smem + loff), 0, 0);
#else
            *(AS3 v4i*)((AS3 char*)smem + loff) = *(const v4i*)(gsrc + goff);
#endif
        }
#if defined(HAVE_ASYNC_TO_LDS)
#  if __has_builtin(__builtin_amdgcn_s_wait_asynccnt)
        __builtin_amdgcn_s_wait_asynccnt(0);
#  else
        asm volatile("s_wait_asynccnt 0x0" ::: "memory");
#  endif
#endif
        __syncthreads();
    }

    const int lane  = threadIdx.x & 31;
    const int wave  = threadIdx.x >> 5;
    const int mBase = blockIdx.x * 32 + (wave >> 2) * 16;
    const int nBase = (wave & 3) * 64;

    v8f acc[4] = {};
    const int rA    = mBase + (lane & 15);
    const int kHalf = (lane >> 4) << 3;  // 0 or 8
    const __bf16* aRow = A + (size_t)rA * K1 + kHalf;

#pragma unroll
    for (int kb = 0; kb < K1; kb += 32) {
        v16bf a = load_frag16(aRow + kb);
#pragma unroll
        for (int j = 0; j < 4; ++j) {
            const int cN = nBase + j * 16 + (lane & 15);
            v16bf b = load_frag16_lds(Bs + cN * BROW_PAD + kb + kHalf);
            acc[j] = __builtin_amdgcn_wmma_f32_16x16x32_bf16(
                false, a, false, b, (short)0, acc[j], false, false);
        }
    }

    // Epilogue: D element (VGPR r, lane l) = (M = r + 8*(l>=16), N = l&15)
    const int nLane = lane & 15;
    const int mOff  = (lane >> 4) << 3;
#pragma unroll
    for (int j = 0; j < 4; ++j) {
        const int col  = nBase + j * 16 + nLane;
        const float bv = b1[col];
#pragma unroll
        for (int r = 0; r < 8; ++r) {
            float v = acc[j][r] + bv;
            v = v > 0.0f ? v : 0.0f;
            h1[(size_t)(mBase + mOff + r) * H_FEATS + col] = f32_to_bf16(v);
        }
    }
}

// ---------------------------------------------------------------------------
// 5) Layer-2 GEMM (pre-aggregation transform, exact since mean is linear):
//    y[200000,16] = h1 @ [W_self2 | W_neigh2]; cols 0-7 self, 8-15 neigh.
//    Bt2 is 8KB -> L2-resident; direct global loads are fine here.
// ---------------------------------------------------------------------------
__global__ void gemm2_kernel(const __bf16* __restrict__ h1,
                             const __bf16* __restrict__ Bt2,
                             float* __restrict__ y) {
    const int lane  = threadIdx.x & 31;
    const int wave  = threadIdx.x >> 5;
    const int mBase = blockIdx.x * 128 + wave * 16;

    int rA = mBase + (lane & 15);
    if (rA >= N_DST0) rA = N_DST0 - 1;           // clamp loads; stores guarded
    const int kHalf = (lane >> 4) << 3;
    const int cN    = lane & 15;

    v8f acc = {};
    const __bf16* aRow = h1 + (size_t)rA * H_FEATS + kHalf;
    const __bf16* bRow = Bt2 + (size_t)cN * H_FEATS + kHalf;
#pragma unroll
    for (int kb = 0; kb < H_FEATS; kb += 32) {
        v16bf a = load_frag16(aRow + kb);
        v16bf b = load_frag16(bRow + kb);
        acc = __builtin_amdgcn_wmma_f32_16x16x32_bf16(
            false, a, false, b, (short)0, acc, false, false);
    }

    const int mOff = (lane >> 4) << 3;
#pragma unroll
    for (int r = 0; r < 8; ++r) {
        int row = mBase + mOff + r;
        if (row < N_DST0) y[(size_t)row * N2 + cN] = acc[r];
    }
}

// ---------------------------------------------------------------------------
// 6) Layer-2 edge scatter on transformed 8-dim messages (8 lanes per edge)
// ---------------------------------------------------------------------------
__global__ void edge2_kernel(const int* __restrict__ src,
                             const int* __restrict__ dst,
                             const float* __restrict__ y,
                             float* __restrict__ agg,
                             float* __restrict__ deg) {
    unsigned t = blockIdx.x * blockDim.x + threadIdx.x;
    int e = t >> 3, c = t & 7;
    if (e >= E1) return;
    int s = src[e], d = dst[e];
    if (c == 0) atomicAdd(&deg[d], 1.0f);
    atomicAdd(&agg[(size_t)d * NUM_CLASSES + c],
              y[(size_t)s * N2 + NUM_CLASSES + c]);   // neigh half (cols 8-15)
}

// ---------------------------------------------------------------------------
// 7) Final: out[n][c] = y_self[n][c] + agg1[n][c]/max(deg,1) + b2[c]
// ---------------------------------------------------------------------------
__global__ void final_kernel(const float* __restrict__ y,
                             const float* __restrict__ agg,
                             const float* __restrict__ deg,
                             const float* __restrict__ b2,
                             float* __restrict__ out) {
    unsigned t = blockIdx.x * blockDim.x + threadIdx.x;
    if (t >= (unsigned)(N_DST1 * NUM_CLASSES)) return;
    int n = t >> 3, c = t & 7;
    float inv = 1.0f / fmaxf(deg[n], 1.0f);
    out[t] = y[(size_t)n * N2 + c] + agg[t] * inv + b2[c];
}

// ---------------------------------------------------------------------------
// Launch
// ---------------------------------------------------------------------------
extern "C" void kernel_launch(void* const* d_in, const int* in_sizes, int n_in,
                              void* d_out, int out_size, void* d_ws, size_t ws_size,
                              hipStream_t stream) {
    const float* x    = (const float*)d_in[0];
    const int*   src0 = (const int*)  d_in[1];
    const int*   dst0 = (const int*)  d_in[2];
    const int*   src1 = (const int*)  d_in[3];
    const int*   dst1 = (const int*)  d_in[4];
    const float* emb  = (const float*)d_in[7];
    const float* Ws1  = (const float*)d_in[8];
    const float* Wn1  = (const float*)d_in[9];
    const float* b1   = (const float*)d_in[10];
    const float* Ws2  = (const float*)d_in[11];
    const float* Wn2  = (const float*)d_in[12];
    const float* b2   = (const float*)d_in[13];

    char* ws = (char*)d_ws;
    auto alloc = [&](size_t bytes) {
        char* p = ws;
        ws += (bytes + 255) & ~(size_t)255;
        return p;
    };
    float*  h    = (float*) alloc((size_t)N_SRC0 * IN_FEATS * 4);
    float*  agg0 = (float*) alloc((size_t)N_DST0 * IN_FEATS * 4);
    float*  deg0 = (float*) alloc((size_t)N_DST0 * 4);
    __bf16* A1   = (__bf16*)alloc((size_t)N_DST0 * K1 * 2);
    __bf16* h1   = (__bf16*)alloc((size_t)N_DST0 * H_FEATS * 2);
    __bf16* Wt1  = (__bf16*)alloc((size_t)K1 * H_FEATS * 2);
    __bf16* Wt2  = (__bf16*)alloc((size_t)N2 * H_FEATS * 2);
    float*  yb   = (float*) alloc((size_t)N_DST0 * N2 * 4);
    float*  agg1 = (float*) alloc((size_t)N_DST1 * NUM_CLASSES * 4);
    float*  deg1 = (float*) alloc((size_t)N_DST1 * 4);

    (void)hipMemsetAsync(agg0, 0, (size_t)N_DST0 * IN_FEATS * 4, stream);
    (void)hipMemsetAsync(deg0, 0, (size_t)N_DST0 * 4, stream);
    (void)hipMemsetAsync(agg1, 0, (size_t)N_DST1 * NUM_CLASSES * 4, stream);
    (void)hipMemsetAsync(deg1, 0, (size_t)N_DST1 * 4, stream);

    gate_kernel   <<<(N_SRC0 * 32) / 256, 256, 0, stream>>>(x, emb, h);
    edge1_kernel  <<<((unsigned)E0 * 32) / 256, 256, 0, stream>>>(src0, dst0, h, agg0, deg0);
    prep_w1_kernel<<<(K1 * H_FEATS) / 256, 256, 0, stream>>>(Ws1, Wn1, Wt1);
    prep_w2_kernel<<<(N2 * H_FEATS) / 256, 256, 0, stream>>>(Ws2, Wn2, Wt2);
    build_a1_kernel<<<(N_DST0 * 32) / 256, 256, 0, stream>>>(h, agg0, deg0, A1);
    gemm1_kernel  <<<N_DST0 / 32, 256, LDS1_BYTES, stream>>>(A1, Wt1, b1, h1);
    gemm2_kernel  <<<(N_DST0 + 127) / 128, 256, 0, stream>>>(h1, Wt2, yb);
    edge2_kernel  <<<((unsigned)E1 * 8) / 256, 256, 0, stream>>>(src1, dst1, yb, agg1, deg1);
    final_kernel  <<<(N_DST1 * NUM_CLASSES + 255) / 256, 256, 0, stream>>>(yb, agg1, deg1, b2, (float*)d_out);
}
